// SSIMLoss_60026462929369
// MI455X (gfx1250) — matrix operand order repaired and checked by reading, not compiled, not run on hardware
//
#include <hip/hip_runtime.h>
#include <hip/hip_bf16.h>

// ---------------------------------------------------------------------------
// SSIM loss, MI455X (gfx1250, wave32).
// 5 depthwise 11x11 Gaussian convs done as separable banded matmuls on the
// WMMA pipe: per 16x16 tile, 2x v_wmma_f32_16x16x32_f16 (vertical) +
// 1x (horizontal) per conv = 15 WMMAs/tile. f16 inputs, f32 accumulate.
// LDS: x1/x2 halo patch (32x144, column-major, pitch 36 halves => 18-bank
// stride, conflict-free) + per-wave 16x32 f16 bounce buffer for the
// D(col-striped) -> A(row-striped) relayout between the two passes.
// ---------------------------------------------------------------------------

typedef __attribute__((ext_vector_type(16))) _Float16 v16h;
typedef __attribute__((ext_vector_type(8)))  _Float16 v8h;
typedef __attribute__((ext_vector_type(4)))  _Float16 v4h;
typedef __attribute__((ext_vector_type(8)))  float    v8f;

#define SSIM_C1 0.0001f
#define SSIM_C2 0.0009f
#define N_PIX   12582912.0   // 16*3*512*512

union U16 { v16h v; v8h o[2]; v4h q[4]; };

// Gaussian tap g[t] = exp(-(t-5)^2/4.5)/sum, t in [0,10]; sum = 3.7592342
__device__ __forceinline__ float gauss_tap(int t) {
  float d = (float)(t - 5);
  return __expf(-d * d * (1.0f / 4.5f)) * 0.26601172f;
}
// Banded matrix entry: tap index t = k - lo - 3 (K dim spans +-8 halo, taps +-5)
__device__ __forceinline__ _Float16 band(int lo, int k) {
  int t = k - lo - 3;
  float g = (t >= 0 && t <= 10) ? gauss_tap(t) : 0.0f;
  return (_Float16)g;
}

__global__ void ssim_init_ws(double* ws) { ws[0] = 0.0; }

__global__ void ssim_finalize(const double* __restrict__ ws, float* __restrict__ out) {
  out[0] = 1.0f - (float)(ws[0] * (1.0 / N_PIX));
}

__global__ __launch_bounds__(256) void ssim_main(const float* __restrict__ img1,
                                                 const float* __restrict__ img2,
                                                 double* __restrict__ ws) {
  // Column-major f16 patches: 144 cols x 32 rows, pitch 36 halves (72B).
  __shared__ __align__(16) _Float16 ldsX1[144 * 36];
  __shared__ __align__(16) _Float16 ldsX2[144 * 36];
  // Per-wave 16x32 f16 Out_v bounce buffer (row-major, pitch 32 halves).
  __shared__ __align__(16) _Float16 ldsV[8][16 * 32];

  const int b       = blockIdx.x;
  const int plane   = b >> 7;        // 48 planes (N*C)
  const int rem     = b & 127;
  const int tileRow = rem >> 2;      // 32 tile-rows
  const int colGrp  = rem & 3;       // 4 groups of 128 cols
  const int r0      = tileRow << 4;
  const int cwg     = colGrp << 7;

  const float* p1 = img1 + (size_t)plane * 262144u;
  const float* p2 = img2 + (size_t)plane * 262144u;

  // ---- Stage halo patch (rows r0-8..r0+23, cols cwg-8..cwg+135), zero-pad OOB.
  for (int idx = threadIdx.x; idx < 32 * 144; idx += 256) {
    int cc = idx % 144;              // consecutive lanes -> consecutive cols (coalesced)
    int k  = idx / 144;
    int gr = r0 - 8 + k;
    int gc = cwg - 8 + cc;
    float a = 0.0f, c = 0.0f;
    if (gr >= 0 && gr < 512 && gc >= 0 && gc < 512) {
      size_t o = (size_t)gr * 512 + gc;
      a = p1[o]; c = p2[o];
    }
    ldsX1[cc * 36 + k] = (_Float16)a;
    ldsX2[cc * 36 + k] = (_Float16)c;
  }
  __syncthreads();

  const int lane = threadIdx.x & 31;
  const int w    = threadIdx.x >> 5;   // wave id: tile column within group
  const int hi   = lane >> 4;          // lane half (WMMA register layouts)
  const int l15  = lane & 15;

  // ---- Gv in 16-bit A layout (16x32): VGPR v holds K=2*koff+2v(+1),
  //      koff = hi*8. Row M = l15.
  v16h aGv;
  {
    int koff = hi << 3;
#pragma unroll
    for (int v = 0; v < 8; ++v) {
      int kb = (koff << 1) + (v << 1);
      aGv[2 * v]     = band(l15, kb);
      aGv[2 * v + 1] = band(l15, kb + 1);
    }
  }
  // ---- Gh in 16-bit B layout (32x16): VGPR v<4 holds K = hi*8 + 2v(+1), v>=4 holds K = 16 + hi*8 + 2(v-4)(+1), col N = l15.
  v16h bGh;
  {
    int kb0 = hi << 4;
#pragma unroll
    for (int v = 0; v < 8; ++v) {
      int k = ((v & 4) << 2) + (kb0 >> 1) + ((v & 3) << 1);
      bGh[2 * v]     = band(l15, k);
      bGh[2 * v + 1] = band(l15, k + 1);
    }
  }

  _Float16* vbuf = &ldsV[w][0];

  // B-matrix (32x16) build from column-major LDS patch: lane's 16 K-halves are
  // contiguous down one column -> 4x ds_load_b64 (8B aligned: pitch 72B).
  auto loadB = [&](const _Float16* src, int h) -> v16h {
    int cc   = (w << 4) + (h << 4) + l15;       // patch col = tile col (c0-8) + h*16 + N
    int base = cc * 36 + (hi << 4);             // K start 0 or 16
    U16 u;
    u.q[0] = *(const v4h*)(src + base + 0);
    u.q[1] = *(const v4h*)(src + base + 4);
    u.q[2] = *(const v4h*)(src + base + 8);
    u.q[3] = *(const v4h*)(src + base + 12);
    return u.v;
  };

  v8f cz = {};
  v8f D[5];

#pragma unroll
  for (int c = 0; c < 5; ++c) {
    // ---- Vertical pass: Out_v[16x16] = Gv x Xpatch for both 16-col halves.
#pragma unroll
    for (int h = 0; h < 2; ++h) {
      v16h bx;
      if (c == 0)      bx = loadB(ldsX1, h);
      else if (c == 1) bx = loadB(ldsX2, h);
      else if (c == 2) { v16h t = loadB(ldsX1, h); bx = t * t; }
      else if (c == 3) { v16h t = loadB(ldsX2, h); bx = t * t; }
      else             { v16h t = loadB(ldsX1, h); v16h s = loadB(ldsX2, h); bx = t * s; }

      v8f dv = __builtin_amdgcn_wmma_f32_16x16x32_f16(
          false, aGv, false, bx, (short)0, cz, false, false);

      // D layout -> row-major f16 in bounce buffer: VGPR m is row m + hi*8, col = h*16+l15.
#pragma unroll
      for (int m = 0; m < 8; ++m) {
        int row = m + (hi << 3);
        vbuf[row * 32 + (h << 4) + l15] = (_Float16)dv[m];
      }
    }

    // ---- Reload Out_v as 16-bit A (16x32): lane row l15, K chunks {koff2..+7, 16+koff2..+7},
    //      koff2 = hi*8 -> two 16B-aligned ds_load_b128. Same-wave DS ops are in-order.
    U16 u;
    {
      int koff2 = hi << 3;
      u.o[0] = *(const v8h*)(vbuf + l15 * 32 + koff2);
      u.o[1] = *(const v8h*)(vbuf + l15 * 32 + 16 + koff2);
    }
    // ---- Horizontal pass: Out[16x16] = Out_v x Gh.
    D[c] = __builtin_amdgcn_wmma_f32_16x16x32_f16(
        false, u.v, false, bGh, (short)0, cz, false, false);
  }

  // ---- Elementwise SSIM map on the 8 accumulator elements per lane.
  float acc = 0.0f;
#pragma unroll
  for (int m = 0; m < 8; ++m) {
    float mu1 = D[0][m], mu2 = D[1][m];
    float e11 = D[2][m], e22 = D[3][m], e12 = D[4][m];
    float mu1sq = mu1 * mu1, mu2sq = mu2 * mu2, mu12 = mu1 * mu2;
    float s1  = e11 - mu1sq;
    float s2  = e22 - mu2sq;
    float s12 = e12 - mu12;
    float num = (2.0f * mu12 + SSIM_C1) * (2.0f * s12 + SSIM_C2);
    float den = (mu1sq + mu2sq + SSIM_C1) * (s1 + s2 + SSIM_C2);
    acc += num / den;
  }

  // ---- Wave32 reduction + one f64 atomic per wave.
#pragma unroll
  for (int off = 16; off > 0; off >>= 1) acc += __shfl_xor(acc, off, 32);
  if (lane == 0) atomicAdd(ws, (double)acc);
}

extern "C" void kernel_launch(void* const* d_in, const int* in_sizes, int n_in,
                              void* d_out, int out_size, void* d_ws, size_t ws_size,
                              hipStream_t stream) {
  (void)in_sizes; (void)n_in; (void)out_size; (void)ws_size;
  const float* img1 = (const float*)d_in[0];
  const float* img2 = (const float*)d_in[1];
  double* ws  = (double*)d_ws;
  float*  out = (float*)d_out;

  ssim_init_ws<<<1, 1, 0, stream>>>(ws);
  // 48 planes * 32 tile-rows * 4 col-groups = 6144 workgroups, 256 thr (8 waves).
  ssim_main<<<dim3(6144), dim3(256), 0, stream>>>(img1, img2, ws);
  ssim_finalize<<<1, 1, 0, stream>>>(ws, out);
}
